// SourceModel_61770219651346
// MI455X (gfx1250) — matrix operand order repaired and checked by reading, not compiled, not run on hardware
//
#include <hip/hip_runtime.h>
#include <hip/hip_bf16.h>

// GNN message-passing block for MI455X (gfx1250, wave32, WMMA).
// Pipeline: pack weights -> edge pass 1 (msg + sum/sumsq atomics) ->
// finalize mean/std -> edge pass 2 (recompute msg, z^3/z^4 atomics) ->
// finalize skew/kurt -> node MLP (160x160, 160x16 WMMA) -> batchnorm.

#define N_NODES 50000
#define N_EDGES 1600000
#define LDIM    16

typedef __attribute__((ext_vector_type(16))) _Float16 v16h;
typedef __attribute__((ext_vector_type(8)))  _Float16 v8h;
typedef __attribute__((ext_vector_type(8)))  float    v8f;

// ---- workspace layout (float offsets) ----
static const size_t SUM_OFF   = 0;                                  // N*32
static const size_t SUMSQ_OFF = (size_t)N_NODES * 32;               // N*32
static const size_t SKEW_OFF  = (size_t)N_NODES * 64;               // N*32
static const size_t KURT_OFF  = (size_t)N_NODES * 96;               // N*32
static const size_t CNT_OFF   = (size_t)N_NODES * 128;              // N
static const size_t OUTP_OFF  = CNT_OFF + N_NODES;                  // N*16
static const size_t BN_OFF    = OUTP_OFF + (size_t)N_NODES * 16;    // 32
static const size_t WS_FLOATS = BN_OFF + 32;                        // ~7.25M floats
// packed f16 weight fragments (offsets in halves, after WS_FLOATS floats)
static const size_t W1A_P = 0;      // 2 frags * 512
static const size_t W2A_P = 1024;   // 2 frags * 512
static const size_t W1B_P = 2048;   // 50 frags * 512
static const size_t W2B_P = 27648;  // 5 frags * 512

#define LDS_FENCE() asm volatile("s_wait_dscnt 0" ::: "memory")

__device__ __forceinline__ v8f wmma16(v16h a, v16h b, v8f c) {
  return __builtin_amdgcn_wmma_f32_16x16x32_f16(false, a, false, b,
                                                (short)0, c, false, false);
}

__device__ __forceinline__ float lk(float x, float s) {
  return (x >= 0.0f) ? x : s * x;
}

// ---- zero accumulators (sum/sumsq/skew/kurt/cnt + bn) ----
__global__ void zero_ws_kernel(float* __restrict__ ws) {
  size_t id = (size_t)blockIdx.x * blockDim.x + threadIdx.x;
  const size_t main_n = CNT_OFF + N_NODES;
  if (id < main_n)               ws[id] = 0.0f;
  else if (id < main_n + 32)     ws[BN_OFF + (id - main_n)] = 0.0f;
}

// ---- pack W[k][n] (row-major KxN f32) into WMMA B-fragment order, f16 ----
// frag (t,b): lane L holds column n = t*16 + (L&15), halves h -> k = b*32 + (h>>3)*16 + (L>>4)*8 + (h&7)
__global__ void pack_w_kernel(const float* __restrict__ W, _Float16* __restrict__ dst,
                              int Nn, int kblocks, int total) {
  int id = blockIdx.x * blockDim.x + threadIdx.x;
  if (id >= total) return;
  int h    = id & 15;
  int lane = (id >> 4) & 31;
  int tb   = id >> 9;
  int b    = tb % kblocks;
  int t    = tb / kblocks;
  int k    = b * 32 + ((h >> 3) << 4) + ((lane >> 4) << 3) + (h & 7);
  int n    = t * 16 + (lane & 15);
  dst[id] = (_Float16)W[(size_t)k * Nn + n];
}

// ---- edge MLP pass: 1 wave = 16 edges; 2 chained 32x32 GEMMs via WMMA ----
template <int PASS>
__global__ __launch_bounds__(256)
void edge_pass_kernel(const float* __restrict__ x_t,
                      const float* __restrict__ edge_attr,
                      const int*   __restrict__ src,
                      const int*   __restrict__ tgt,
                      const float* __restrict__ b1a,
                      const float* __restrict__ b2a,
                      const _Float16* __restrict__ wp,
                      float* __restrict__ ws) {
  __shared__ __align__(16) _Float16 stg[8][16 * 40];   // per-wave 16x32 staging, padded
  const int w    = threadIdx.x >> 5;
  const int lane = threadIdx.x & 31;
  const int hg   = lane >> 4;
  const int m    = lane & 15;         // A row / C column
  const int col  = m;
  const size_t e0 = ((size_t)blockIdx.x * 8 + w) * 16;

  // A fragment: rows = edges, K 0..15 = x_t[tgt], K 16..31 = edge_attr
  const int    te = tgt[e0 + m];
  const float* xr = x_t       + (size_t)te * LDIM + hg * 8;
  const float* er = edge_attr + (e0 + m)  * LDIM + hg * 8;
  v16h a;
#pragma unroll
  for (int i = 0; i < 8; ++i) { a[i] = (_Float16)xr[i]; a[8 + i] = (_Float16)er[i]; }

  // layer 1: h = leaky(A @ W1a + b1a, 0.1)
  v8f c0, c1;
  { float bb0 = b1a[col], bb1 = b1a[16 + col];
#pragma unroll
    for (int v = 0; v < 8; ++v) { c0[v] = bb0; c1[v] = bb1; } }
  v16h bf0 = *(const v16h*)(wp + W1A_P + (size_t)(0 * 32 + lane) * 16);
  v16h bf1 = *(const v16h*)(wp + W1A_P + (size_t)(1 * 32 + lane) * 16);
  c0 = wmma16(a, bf0, c0);
  c1 = wmma16(a, bf1, c1);

  // C -> A relayout through LDS (leaky applied in f32 before f16 narrowing)
#pragma unroll
  for (int v = 0; v < 8; ++v) {
    int r = hg * 8 + v;
    stg[w][r * 40 + col]      = (_Float16)lk(c0[v], 0.1f);
    stg[w][r * 40 + 16 + col] = (_Float16)lk(c1[v], 0.1f);
  }
  LDS_FENCE();
  v16h a2;
  { const v8h lo = *(const v8h*)&stg[w][m * 40 + hg * 8];
    const v8h hi = *(const v8h*)&stg[w][m * 40 + 16 + hg * 8];
#pragma unroll
    for (int i = 0; i < 8; ++i) { a2[i] = lo[i]; a2[8 + i] = hi[i]; } }

  // layer 2: msg = h @ W2a + b2a
  v8f d0, d1;
  { float bb0 = b2a[col], bb1 = b2a[16 + col];
#pragma unroll
    for (int v = 0; v < 8; ++v) { d0[v] = bb0; d1[v] = bb1; } }
  v16h bg0 = *(const v16h*)(wp + W2A_P + (size_t)(0 * 32 + lane) * 16);
  v16h bg1 = *(const v16h*)(wp + W2A_P + (size_t)(1 * 32 + lane) * 16);
  d0 = wmma16(a2, bg0, d0);
  d1 = wmma16(a2, bg1, d1);

  int sidx[8];
#pragma unroll
  for (int v = 0; v < 8; ++v) sidx[v] = src[e0 + hg * 8 + v];

  if (PASS == 1) {
#pragma unroll
    for (int v = 0; v < 8; ++v) {
      size_t r0 = (size_t)sidx[v] * 32 + col;
      float x0 = d0[v], x1 = d1[v];
      atomicAdd(&ws[SUM_OFF   + r0],      x0);
      atomicAdd(&ws[SUMSQ_OFF + r0],      x0 * x0);
      atomicAdd(&ws[SUM_OFF   + r0 + 16], x1);
      atomicAdd(&ws[SUMSQ_OFF + r0 + 16], x1 * x1);
    }
    if (lane < 16) atomicAdd(&ws[CNT_OFF + src[e0 + lane]], 1.0f);
  } else {
#pragma unroll
    for (int v = 0; v < 8; ++v) {
      size_t r0 = (size_t)sidx[v] * 32 + col;
      float mu0 = ws[SUM_OFF + r0],      sd0 = ws[SUMSQ_OFF + r0];
      float mu1 = ws[SUM_OFF + r0 + 16], sd1 = ws[SUMSQ_OFF + r0 + 16];
      float z0 = (d0[v] - mu0) / sd0;
      float z1 = (d1[v] - mu1) / sd1;
      atomicAdd(&ws[SKEW_OFF + r0],      z0 * z0 * z0);
      atomicAdd(&ws[KURT_OFF + r0],      z0 * z0 * z0 * z0);
      atomicAdd(&ws[SKEW_OFF + r0 + 16], z1 * z1 * z1);
      atomicAdd(&ws[KURT_OFF + r0 + 16], z1 * z1 * z1 * z1);
    }
  }
}

// ---- mean/std finalize (in place: sum->mean, sumsq->std) ----
__global__ void finalize_moments1(float* __restrict__ ws) {
  size_t id = (size_t)blockIdx.x * blockDim.x + threadIdx.x;
  if (id >= (size_t)N_NODES * 32) return;
  size_t i = id >> 5;
  float denom = fmaxf(ws[CNT_OFF + i], 1.0f);
  float mean  = ws[SUM_OFF + id] / denom;
  float ex2   = ws[SUMSQ_OFF + id] / denom;
  float var   = ex2 - mean * mean;
  var = (var >= 0.0f) ? var : 0.01f * var;
  ws[SUM_OFF + id]   = mean;
  ws[SUMSQ_OFF + id] = sqrtf(var + 1e-6f);
}

// ---- skew/kurt finalize with nan_to_num ----
__global__ void finalize_moments2(float* __restrict__ ws) {
  size_t id = (size_t)blockIdx.x * blockDim.x + threadIdx.x;
  if (id >= (size_t)N_NODES * 32) return;
  size_t i = id >> 5;
  float denom = fmaxf(ws[CNT_OFF + i], 1.0f);
  float s = ws[SKEW_OFF + id] / denom;
  float k = ws[KURT_OFF + id] / denom;
  if (s != s) s = 0.0f; s = fminf(fmaxf(s, -3.4e38f), 3.4e38f);
  if (k != k) k = 0.0f; k = fminf(fmaxf(k, -3.4e38f), 3.4e38f);
  ws[SKEW_OFF + id] = s;
  ws[KURT_OFF + id] = k;
}

__device__ __forceinline__ const float* feat_ptr(int f, int i,
    const float* xs, const float* mean, const float* stdv,
    const float* skw, const float* krt, const float* u) {
  if (f < 16)  return xs   + (size_t)i * 16 + f;
  if (f < 48)  return mean + (size_t)i * 32 + (f - 16);
  if (f < 80)  return stdv + (size_t)i * 32 + (f - 48);
  if (f < 112) return skw  + (size_t)i * 32 + (f - 80);
  if (f < 144) return krt  + (size_t)i * 32 + (f - 112);
  return u + (f - 144);
}

// ---- node MLP: 1 wave = 16 nodes; [16x160]@[160x160] then @[160x16] ----
__global__ __launch_bounds__(256)
void node_mlp_kernel(const float* __restrict__ xs,
                     const float* __restrict__ u,
                     const float* __restrict__ b1b,
                     const float* __restrict__ b2b,
                     const _Float16* __restrict__ wp,
                     float* __restrict__ ws) {
  __shared__ __align__(16) _Float16 stg[8][16 * 168];  // per-wave 16x160, padded
  const int w    = threadIdx.x >> 5;
  const int lane = threadIdx.x & 31;
  const int hg   = lane >> 4;
  const int m    = lane & 15;
  const int col  = m;
  const int tile = blockIdx.x * 8 + w;
  if (tile * 16 >= N_NODES) return;   // per-wave LDS only; no block barrier used
  const int i = tile * 16 + m;

  const float* mean = ws + SUM_OFF;
  const float* stdv = ws + SUMSQ_OFF;
  const float* skw  = ws + SKEW_OFF;
  const float* krt  = ws + KURT_OFF;

  // A fragments for K = 160 (5 blocks of 32)
  v16h af[5];
#pragma unroll
  for (int b = 0; b < 5; ++b) {
    int f0 = b * 32 + hg * 8;
    const float* p0 = feat_ptr(f0,      i, xs, mean, stdv, skw, krt, u);
    const float* p1 = feat_ptr(f0 + 16, i, xs, mean, stdv, skw, krt, u);
#pragma unroll
    for (int j = 0; j < 8; ++j) { af[b][j] = (_Float16)p0[j]; af[b][8 + j] = (_Float16)p1[j]; }
  }

  // layer 1: h2 = leaky(h_cat @ W1b + b1b, 0.1)  (10 N-tiles x 5 K-steps)
#pragma unroll
  for (int t = 0; t < 10; ++t) {
    v8f acc;
    float bv = b1b[t * 16 + col];
#pragma unroll
    for (int v = 0; v < 8; ++v) acc[v] = bv;
#pragma unroll
    for (int b = 0; b < 5; ++b) {
      v16h bf = *(const v16h*)(wp + W1B_P + (size_t)((t * 5 + b) * 32 + lane) * 16);
      acc = wmma16(af[b], bf, acc);
    }
#pragma unroll
    for (int v = 0; v < 8; ++v)
      stg[w][(hg * 8 + v) * 168 + t * 16 + col] = (_Float16)lk(acc[v], 0.1f);
  }
  LDS_FENCE();

  // layer 2: out = h2 @ W2b + b2b
  v8f o;
  { float bv2 = b2b[col];
#pragma unroll
    for (int v = 0; v < 8; ++v) o[v] = bv2; }
#pragma unroll
  for (int b = 0; b < 5; ++b) {
    v16h a2;
    const v8h lo = *(const v8h*)&stg[w][m * 168 + b * 32 + hg * 8];
    const v8h hi = *(const v8h*)&stg[w][m * 168 + b * 32 + 16 + hg * 8];
#pragma unroll
    for (int j = 0; j < 8; ++j) { a2[j] = lo[j]; a2[8 + j] = hi[j]; }
    v16h bf = *(const v16h*)(wp + W2B_P + (size_t)(b * 32 + lane) * 16);
    o = wmma16(a2, bf, o);
  }

  // store pre-BN output + batch stats (reduce across lane halves first)
  float ps = 0.0f, pss = 0.0f;
#pragma unroll
  for (int v = 0; v < 8; ++v) {
    int row = tile * 16 + hg * 8 + v;
    float x = o[v];
    ws[OUTP_OFF + (size_t)row * 16 + col] = x;
    ps += x;
    pss += x * x;
  }
  ps  += __shfl_xor(ps, 16, 32);
  pss += __shfl_xor(pss, 16, 32);
  if (lane < 16) {
    atomicAdd(&ws[BN_OFF + col], ps);
    atomicAdd(&ws[BN_OFF + 16 + col], pss);
  }
}

// ---- batchnorm finalize ----
__global__ void bn_final_kernel(const float* __restrict__ gamma,
                                const float* __restrict__ beta,
                                const float* __restrict__ ws,
                                float* __restrict__ out) {
  size_t id = (size_t)blockIdx.x * blockDim.x + threadIdx.x;
  if (id >= (size_t)N_NODES * 16) return;
  int j = (int)(id & 15);
  const float inv_n = 1.0f / (float)N_NODES;
  float mu  = ws[BN_OFF + j] * inv_n;
  float msq = ws[BN_OFF + 16 + j] * inv_n;
  float var = msq - mu * mu;
  out[id] = gamma[j] * (ws[OUTP_OFF + id] - mu) * rsqrtf(var + 1e-5f) + beta[j];
}

extern "C" void kernel_launch(void* const* d_in, const int* in_sizes, int n_in,
                              void* d_out, int out_size, void* d_ws, size_t ws_size,
                              hipStream_t stream) {
  const float* x_s  = (const float*)d_in[0];
  const float* x_t  = (const float*)d_in[1];
  const float* eat  = (const float*)d_in[2];
  const float* u    = (const float*)d_in[3];
  const float* W1a  = (const float*)d_in[4];
  const float* b1a  = (const float*)d_in[5];
  const float* W2a  = (const float*)d_in[6];
  const float* b2a  = (const float*)d_in[7];
  const float* W1b  = (const float*)d_in[8];
  const float* b1b  = (const float*)d_in[9];
  const float* W2b  = (const float*)d_in[10];
  const float* b2b  = (const float*)d_in[11];
  const float* gamv = (const float*)d_in[12];
  const float* betv = (const float*)d_in[13];
  const int*   ei   = (const int*)d_in[14];
  const int*   src  = ei;
  const int*   tgt  = ei + N_EDGES;

  float*    ws = (float*)d_ws;
  _Float16* wp = (_Float16*)((char*)d_ws + WS_FLOATS * sizeof(float));

  { size_t total = CNT_OFF + N_NODES + 32;
    zero_ws_kernel<<<(unsigned)((total + 255) / 256), 256, 0, stream>>>(ws); }

  pack_w_kernel<<<(1024 + 255) / 256, 256, 0, stream>>>(W1a, wp + W1A_P, 32, 1, 1024);
  pack_w_kernel<<<(1024 + 255) / 256, 256, 0, stream>>>(W2a, wp + W2A_P, 32, 1, 1024);
  pack_w_kernel<<<(25600 + 255) / 256, 256, 0, stream>>>(W1b, wp + W1B_P, 160, 5, 25600);
  pack_w_kernel<<<(2560 + 255) / 256, 256, 0, stream>>>(W2b, wp + W2B_P, 16, 5, 2560);

  const int edge_blocks = N_EDGES / (16 * 8);  // 12500
  edge_pass_kernel<1><<<edge_blocks, 256, 0, stream>>>(x_t, eat, src, tgt, b1a, b2a, wp, ws);

  finalize_moments1<<<((size_t)N_NODES * 32 + 255) / 256, 256, 0, stream>>>(ws);

  edge_pass_kernel<2><<<edge_blocks, 256, 0, stream>>>(x_t, eat, src, tgt, b1a, b2a, wp, ws);

  finalize_moments2<<<((size_t)N_NODES * 32 + 255) / 256, 256, 0, stream>>>(ws);

  const int node_tiles = N_NODES / 16;  // 3125
  node_mlp_kernel<<<(node_tiles + 7) / 8, 256, 0, stream>>>(x_s, u, b1b, b2b, wp, ws);

  bn_final_kernel<<<((size_t)N_NODES * 16 + 255) / 256, 256, 0, stream>>>(
      gamv, betv, ws, (float*)d_out);
}
